// MSC_7215545057886
// MI455X (gfx1250) — compile-verified, matrix operand order checked by reading, not run on hardware
//
#include <hip/hip_runtime.h>
#include <math.h>

#define B_    2
#define C_    256
#define HW_   32
#define N_    1024
#define NH_   8
#define HD_   32
#define TOPK1 512
#define TOPK2 341
#define SCALE_F 0.17677669529663687f  // 32^-0.5

typedef __attribute__((ext_vector_type(16))) _Float16 v16h;
typedef __attribute__((ext_vector_type(8)))  _Float16 v8h;
typedef __attribute__((ext_vector_type(8)))  float    v8f;

// ---------------- wave32 reductions ----------------
__device__ __forceinline__ float wred_addf(float v){
#pragma unroll
  for (int o = 16; o; o >>= 1) v += __shfl_xor(v, o, 32);
  return v;
}
__device__ __forceinline__ int wred_addi(int v){
#pragma unroll
  for (int o = 16; o; o >>= 1) v += __shfl_xor(v, o, 32);
  return v;
}

// order-preserving float <-> uint key
__device__ __forceinline__ unsigned f2k(float f){
  unsigned u = __float_as_uint(f);
  return (u >> 31) ? ~u : (u | 0x80000000u);
}
__device__ __forceinline__ float k2f(unsigned k){
  unsigned u = (k & 0x80000000u) ? (k & 0x7FFFFFFFu) : ~k;
  return __uint_as_float(u);
}

// kk-th largest key among 32 keys/lane * 32 lanes (uniform binary search)
__device__ __forceinline__ unsigned kth_key(const unsigned* keys, int kk){
  unsigned lo = 0u, hi = 0xFFFFFFFFu, ans = 0u;
#pragma unroll 1
  for (int it = 0; it < 33 && lo <= hi; ++it){
    unsigned mid = lo + ((hi - lo) >> 1);
    int c = 0;
#pragma unroll
    for (int i = 0; i < 32; ++i) c += (keys[i] >= mid) ? 1 : 0;
    c = wred_addi(c);
    if (c >= kk){ ans = mid; if (mid == 0xFFFFFFFFu) break; lo = mid + 1; }
    else        {            if (mid == 0u)          break; hi = mid - 1; }
  }
  return ans;
}

// ---------------- WMMA fragment helpers ----------------
// 16-bit 16x32 fragment striping (ISA 7.12.2): lane m = lane&15,
// halves 0-7 -> K = base..base+7, halves 8-15 -> K = base+16.., base = 8*(lane>=16).
__device__ __forceinline__ const _Float16* frag_ptr(const _Float16* base, int stride, int lane){
  return base + (size_t)(lane & 15) * stride + ((lane >> 4) << 3);
}
__device__ __forceinline__ v16h frag_ld(const _Float16* p0){
  v8h lo = *(const v8h*)(p0);
  v8h hi = *(const v8h*)(p0 + 16);
  v16h r;
#pragma unroll
  for (int i = 0; i < 8; ++i){ r[i] = lo[i]; r[i + 8] = hi[i]; }
  return r;
}
__device__ __forceinline__ v8f wmma32(v16h a, v16h b, v8f c){
  return __builtin_amdgcn_wmma_f32_16x16x32_f16(false, a, false, b,
                                                (short)0, c, false, false);
}

// One wave computes a 16(M) x 64(N) register-blocked tile of A[M,256] @ W[N,256]^T.
// A fragment is reused across 4 WMMAs: 5 fragment loads per 4 WMMAs.
__device__ __forceinline__ void gemm_tile4(const _Float16* __restrict__ A,
                                           const _Float16* __restrict__ W,
                                           int tm, int tn4, int lane, v8f acc[4]){
  const _Float16* ap = frag_ptr(A + (size_t)tm * 16 * C_, C_, lane);
  const _Float16* bp = frag_ptr(W + (size_t)tn4 * 64 * C_, C_, lane);
#pragma unroll
  for (int t = 0; t < 4; ++t) acc[t] = v8f{};
#pragma unroll
  for (int kc = 0; kc < C_; kc += 32){
    v16h a = frag_ld(ap + kc);
#pragma unroll
    for (int t = 0; t < 4; ++t){
      v16h b = frag_ld(bp + (size_t)t * 16 * C_ + kc);
      acc[t] = wmma32(a, b, acc[t]);
    }
  }
}

// ---------------- stage kernels ----------------

// fused 3-scale avgpool (count_include_pad): tap weight = sum of 1/k^2 over windows containing it
__global__ void msc_pool(const float* __restrict__ y, float* __restrict__ pooled){
  int idx = blockIdx.x * 256 + threadIdx.x;           // over B*C*H*W
  if (idx >= B_ * C_ * HW_ * HW_) return;
  int wx = idx & 31, hy = (idx >> 5) & 31, c = (idx >> 10) & 255, b = idx >> 18;
  const float* yb = y + ((size_t)(b * C_ + c) << 10);
  const float w3 = 1.f/9.f, w5 = 1.f/25.f, w7 = 1.f/49.f;
  float acc = 0.f;
#pragma unroll
  for (int dy = -3; dy <= 3; ++dy){
    int yy = hy + dy; if (yy < 0 || yy >= HW_) continue;
#pragma unroll
    for (int dx = -3; dx <= 3; ++dx){
      int xx = wx + dx; if (xx < 0 || xx >= HW_) continue;
      int ay = dy < 0 ? -dy : dy, ax = dx < 0 ? -dx : dx;
      float wt = w7;
      if (ay <= 2 && ax <= 2) wt += w5;
      if (ay <= 1 && ax <= 1) wt += w3;
      acc += yb[yy * HW_ + xx] * wt;
    }
  }
  pooled[(size_t)((b << 10) + (hy << 5) + wx) * C_ + c] = acc;   // [B,N,C]
}

// LayerNorm over C=256, one 8-wave block per token row; f16 output
__global__ void msc_ln(const float* __restrict__ pooled, const float* __restrict__ g,
                       const float* __restrict__ bta, _Float16* __restrict__ yf16){
  int row = blockIdx.x, c = threadIdx.x;
  float v = pooled[(size_t)row * C_ + c];
  __shared__ float sh1[8], sh2[8];
  float s  = wred_addf(v);
  float s2 = wred_addf(v * v);
  if ((c & 31) == 0){ sh1[c >> 5] = s; sh2[c >> 5] = s2; }
  __syncthreads();
  float tot = 0.f, tot2 = 0.f;
#pragma unroll
  for (int i = 0; i < 8; ++i){ tot += sh1[i]; tot2 += sh2[i]; }
  float mu  = tot * (1.f / C_);
  float var = tot2 * (1.f / C_) - mu * mu;
  float inv = rsqrtf(var + 1e-5f);
  yf16[(size_t)row * C_ + c] = (_Float16)((v - mu) * inv * g[c] + bta[c]);
}

// x [B,C,N] fp32  ->  xf16 [B,N,C] f16
__global__ void msc_xpose(const float* __restrict__ x, _Float16* __restrict__ xf16){
  int idx = blockIdx.x * 256 + threadIdx.x;           // (b*N + n)*C + c
  if (idx >= B_ * N_ * C_) return;
  int c = idx & 255, n = (idx >> 8) & 1023, b = idx >> 18;
  xf16[idx] = (_Float16)x[((size_t)(b * C_ + c) << 10) + n];
}

__global__ void msc_cvt(const float* __restrict__ s, _Float16* __restrict__ d, int n){
  int i = blockIdx.x * 256 + threadIdx.x;
  if (i < n) d[i] = (_Float16)s[i];
}

// q = xf @ Wq^T  -> q16 [B,h,n,d] f16   (grid: M/16 x N/64)
__global__ __launch_bounds__(32) void msc_gemm_q(const _Float16* __restrict__ A,
                                                 const _Float16* __restrict__ W,
                                                 _Float16* __restrict__ q16){
  int lane = threadIdx.x;
  v8f acc[4];
  gemm_tile4(A, W, blockIdx.x, blockIdx.y, lane, acc);
  int n = lane & 15, mb = (lane >> 4) << 3;
#pragma unroll
  for (int t = 0; t < 4; ++t){
    int col = blockIdx.y * 64 + t * 16 + n;
    int h = col >> 5, dd = col & 31;
#pragma unroll
    for (int j = 0; j < 8; ++j){
      int row = blockIdx.x * 16 + mb + j;
      int b = row >> 10, nn = row & 1023;
      q16[(size_t)(((b * NH_ + h) << 10) + nn) * HD_ + dd] = (_Float16)acc[t][j];
    }
  }
}

// kv = yf @ Wkv^T -> k16 [B,h,n,d] f16, vT16 [B,h,d,n] f16 (transposed for P@V fragments)
__global__ __launch_bounds__(32) void msc_gemm_kv(const _Float16* __restrict__ A,
                                                  const _Float16* __restrict__ W,
                                                  _Float16* __restrict__ k16,
                                                  _Float16* __restrict__ vT16){
  int lane = threadIdx.x;
  v8f acc[4];
  gemm_tile4(A, W, blockIdx.x, blockIdx.y, lane, acc);
  int n = lane & 15, mb = (lane >> 4) << 3;
#pragma unroll
  for (int t = 0; t < 4; ++t){
    int col = blockIdx.y * 64 + t * 16 + n;
#pragma unroll
    for (int j = 0; j < 8; ++j){
      int row = blockIdx.x * 16 + mb + j;
      int b = row >> 10, nn = row & 1023;
      if (col < C_){
        int h = col >> 5, dd = col & 31;
        k16[(size_t)(((b * NH_ + h) << 10) + nn) * HD_ + dd] = (_Float16)acc[t][j];
      } else {
        int c2 = col - C_, h = c2 >> 5, dd = c2 & 31;
        vT16[(size_t)((b * NH_ + h) * HD_ + dd) * N_ + nn] = (_Float16)acc[t][j];
      }
    }
  }
}

// fused attention: scores (WMMA) -> dual top-k thresholds -> dual softmax -> P@V (WMMA)
// one wave per 16-query tile; 96KB dynamic LDS (CDNA5 allows up to 320KB/WG)
__global__ __launch_bounds__(32) void msc_attn(const _Float16* __restrict__ q16,
                                               const _Float16* __restrict__ k16,
                                               const _Float16* __restrict__ vT16,
                                               const float* __restrict__ p1p,
                                               const float* __restrict__ p2p,
                                               _Float16* __restrict__ ao16){
  extern __shared__ char smem[];
  float*    sc = (float*)smem;                         // 16 x 1024 f32 scores (64KB)
  _Float16* pw = (_Float16*)(smem + 16 * 1024 * 4);    // 16 x 1024 f16 weights (32KB)
  int lane = threadIdx.x;
  int rt = blockIdx.x;                                 // query tile 0..63
  int bh = blockIdx.y;                                 // 0..15 (b*8+h)
  const _Float16* qb = q16  + (size_t)((bh << 10) + rt * 16) * HD_;
  const _Float16* kb = k16  + ((size_t)bh << 10) * HD_;
  const _Float16* vb = vT16 + (size_t)bh * HD_ * N_;
  int n = lane & 15, mb = (lane >> 4) << 3;

  // phase 1: S = q_tile(16x32) . K^T -> 16x1024 scaled scores in LDS
  // A fragment (q) is loop-invariant -> 1 load per WMMA.
  v16h aq = frag_ld(frag_ptr(qb, HD_, lane));
  const _Float16* kfp = frag_ptr(kb, HD_, lane);
#pragma unroll 4
  for (int ct = 0; ct < 64; ++ct){
    v16h bk = frag_ld(kfp + (size_t)ct * 16 * HD_);
    v8f z = {};
    v8f acc = wmma32(aq, bk, z);
#pragma unroll
    for (int j = 0; j < 8; ++j)
      sc[(mb + j) * 1024 + ct * 16 + n] = acc[j] * SCALE_F;
  }
  __syncthreads();

  // phase 2: per row top-k thresholds + combined dual-softmax weights (f16)
  float p1 = p1p[0], p2 = p2p[0];
  for (int m = 0; m < 16; ++m){
    unsigned keys[32]; float ev[32];
#pragma unroll
    for (int i = 0; i < 32; ++i) keys[i] = f2k(sc[m * 1024 + i * 32 + lane]);
    unsigned t1 = kth_key(keys, TOPK1);
    unsigned t2 = kth_key(keys, TOPK2);
    unsigned mk = keys[0];
#pragma unroll
    for (int i = 1; i < 32; ++i) mk = keys[i] > mk ? keys[i] : mk;
#pragma unroll
    for (int o = 16; o; o >>= 1){
      unsigned t = (unsigned)__shfl_xor((int)mk, o, 32);
      mk = t > mk ? t : mk;
    }
    float rowmax = k2f(mk);
    float s1 = 0.f, s2 = 0.f;
#pragma unroll
    for (int i = 0; i < 32; ++i){
      float e = __expf(k2f(keys[i]) - rowmax);
      ev[i] = e;
      if (keys[i] >= t1) s1 += e;
      if (keys[i] >= t2) s2 += e;
    }
    s1 = wred_addf(s1); s2 = wred_addf(s2);
    float r1 = p1 / s1, r2 = p2 / s2;
#pragma unroll
    for (int i = 0; i < 32; ++i){
      float wgt = ev[i] * ((keys[i] >= t1 ? r1 : 0.f) + (keys[i] >= t2 ? r2 : 0.f));
      pw[m * 1024 + i * 32 + lane] = (_Float16)wgt;
    }
  }
  __syncthreads();

  // phase 3: out_tile(16x32) = PW(16x1024) @ V(1024x32), K tiled by 32.
  // A fragment (pw) reused across both d-column tiles: 3 loads per 2 WMMAs.
  int b = bh >> 3, h = bh & 7;
  v8f acc0 = {}, acc1 = {};
  const _Float16* pwp = frag_ptr(pw, N_, lane);
  const _Float16* vp0 = frag_ptr(vb, N_, lane);
  const _Float16* vp1 = frag_ptr(vb + (size_t)16 * N_, N_, lane);
#pragma unroll 4
  for (int kc = 0; kc < 32; ++kc){
    v16h a = frag_ld(pwp + kc * 32);
    acc0 = wmma32(a, frag_ld(vp0 + kc * 32), acc0);
    acc1 = wmma32(a, frag_ld(vp1 + kc * 32), acc1);
  }
#pragma unroll
  for (int j = 0; j < 8; ++j){
    int row = (b << 10) + rt * 16 + mb + j;
    ao16[(size_t)row * C_ + (h << 5) + n]      = (_Float16)acc0[j];
    ao16[(size_t)row * C_ + (h << 5) + 16 + n] = (_Float16)acc1[j];
  }
}

// out = attn_out @ Wproj^T + bproj, written as [B,C,H,W] fp32   (grid: M/16 x N/64)
__global__ __launch_bounds__(32) void msc_gemm_proj(const _Float16* __restrict__ A,
                                                    const _Float16* __restrict__ W,
                                                    const float* __restrict__ bias,
                                                    float* __restrict__ out){
  int lane = threadIdx.x;
  v8f acc[4];
  gemm_tile4(A, W, blockIdx.x, blockIdx.y, lane, acc);
  int n = lane & 15, mb = (lane >> 4) << 3;
#pragma unroll
  for (int t = 0; t < 4; ++t){
    int col = blockIdx.y * 64 + t * 16 + n;
    float bz = bias[col];
#pragma unroll
    for (int j = 0; j < 8; ++j){
      int row = blockIdx.x * 16 + mb + j;
      int b = row >> 10, nn = row & 1023;
      out[((size_t)(b * C_ + col) << 10) + nn] = acc[t][j] + bz;
    }
  }
}

// ---------------- host launcher ----------------
extern "C" void kernel_launch(void* const* d_in, const int* in_sizes, int n_in,
                              void* d_out, int out_size, void* d_ws, size_t ws_size,
                              hipStream_t stream){
  (void)in_sizes; (void)n_in; (void)out_size; (void)ws_size;
  const float* x     = (const float*)d_in[0];
  const float* y     = (const float*)d_in[1];
  const float* Wq    = (const float*)d_in[2];
  const float* Wkv   = (const float*)d_in[3];
  const float* Wproj = (const float*)d_in[4];
  const float* bproj = (const float*)d_in[5];
  const float* ln_g  = (const float*)d_in[6];
  const float* ln_b  = (const float*)d_in[7];
  const float* p1    = (const float*)d_in[8];
  const float* p2    = (const float*)d_in[9];
  float* out = (float*)d_out;

  char* w = (char*)d_ws;
  float*    pooled = (float*)w;      w += (size_t)B_ * N_ * C_ * 4;
  _Float16* yf16   = (_Float16*)w;   w += (size_t)B_ * N_ * C_ * 2;
  _Float16* xf16   = (_Float16*)w;   w += (size_t)B_ * N_ * C_ * 2;
  _Float16* wq16   = (_Float16*)w;   w += (size_t)C_ * C_ * 2;
  _Float16* wkv16  = (_Float16*)w;   w += (size_t)2 * C_ * C_ * 2;
  _Float16* wproj16= (_Float16*)w;   w += (size_t)C_ * C_ * 2;
  _Float16* q16    = (_Float16*)w;   w += (size_t)B_ * NH_ * N_ * HD_ * 2;
  _Float16* k16    = (_Float16*)w;   w += (size_t)B_ * NH_ * N_ * HD_ * 2;
  _Float16* vT16   = (_Float16*)w;   w += (size_t)B_ * NH_ * HD_ * N_ * 2;
  _Float16* ao16   = (_Float16*)w;

  // allow 96KB dynamic LDS for the fused attention kernel (CDNA5 big-LDS)
  (void)hipFuncSetAttribute((const void*)msc_attn,
                            hipFuncAttributeMaxDynamicSharedMemorySize, 96 * 1024);

  msc_pool <<<dim3((B_*C_*HW_*HW_ + 255) / 256), 256, 0, stream>>>(y, pooled);
  msc_ln   <<<dim3(B_*N_), 256, 0, stream>>>(pooled, ln_g, ln_b, yf16);
  msc_xpose<<<dim3((B_*N_*C_ + 255) / 256), 256, 0, stream>>>(x, xf16);
  msc_cvt  <<<dim3((C_*C_   + 255) / 256), 256, 0, stream>>>(Wq,    wq16,    C_*C_);
  msc_cvt  <<<dim3((2*C_*C_ + 255) / 256), 256, 0, stream>>>(Wkv,   wkv16,   2*C_*C_);
  msc_cvt  <<<dim3((C_*C_   + 255) / 256), 256, 0, stream>>>(Wproj, wproj16, C_*C_);

  msc_gemm_q  <<<dim3(B_*N_/16, C_/64),   32, 0, stream>>>(xf16, wq16,  q16);
  msc_gemm_kv <<<dim3(B_*N_/16, 2*C_/64), 32, 0, stream>>>(yf16, wkv16, k16, vT16);

  msc_attn <<<dim3(N_/16, B_*NH_), 32, 96 * 1024, stream>>>(q16, k16, vT16, p1, p2, ao16);

  msc_gemm_proj<<<dim3(B_*N_/16, C_/64), 32, 0, stream>>>(ao16, wproj16, bproj, out);
}
